// HeatrateOption_5927054869110
// MI455X (gfx1250) — matrix operand order repaired and checked by reading, not compiled, not run on hardware
//
#include <hip/hip_runtime.h>
#include <hip/hip_bf16.h>

typedef __attribute__((ext_vector_type(16))) _Float16 v16h;
typedef __attribute__((ext_vector_type(8)))  float    v8f;

#define NB      512
#define BATCH   131072
#define NSTEPS  260
#define TM      64            // paths per block
#define KT16    16            // 512 / 32 (WMMA K chunks)
#define MT4     4             // 64 / 16  (WMMA M tiles)

#define DT_F     (1.0f/260.0f)
#define DRIFT_P  (-2.25961538461e-4f)   // (0.0025 - 0.35^2/2) * DT
#define DRIFT_G  (-6.82692307692e-4f)   // (0.0025 - 0.60^2/2) * DT

// ---------------------------------------------------------------------------
// Prep: repack W2 / W3 into WMMA B-fragment order (f16).
//   B-fragment (16-bit, 32x16):  n = lane % 16 ; k = kt*32 + (e/8)*16 + (lane/16)*8 + e%8
// ---------------------------------------------------------------------------
__global__ void hr_pack(const float* __restrict__ W2, const float* __restrict__ W3,
                        _Float16* __restrict__ w2p, _Float16* __restrict__ w3p)
{
    const int q = blockIdx.x * 256 + threadIdx.x;   // 0 .. 262143
    {   // W2 -> [nt][kt][lane][e] f16
        int e = q & 15, lane = (q >> 4) & 31, kt = (q >> 9) & 15, nt = q >> 13;
        int n = (nt << 4) | (lane & 15);
        int k = (kt << 5) + ((e >> 3) << 4) + ((lane >> 4) << 3) + (e & 7);
        w2p[q] = (_Float16)W2[k * NB + n];
    }
    if (q < 8192) {  // W3 (512x2) zero-padded to 512x16 -> [kt][lane][e]
        int e = q & 15, lane = (q >> 4) & 31, kt = q >> 9;
        int n = lane & 15;
        int k = (kt << 5) + ((e >> 3) << 4) + ((lane >> 4) << 3) + (e & 7);
        w3p[q] = (n < 2) ? (_Float16)W3[k * 2 + n] : (_Float16)0.0f;
    }
}

// ---------------------------------------------------------------------------
// Main persistent-path kernel: 256 threads (8 waves), 64 paths per block.
// Per step:  L1 (VALU, LDS-resident weights) -> L2 GEMM (WMMA, W2 streamed
// from L2 in B-fragment order) -> L3 (WMMA, N padded to 16) -> path update.
// ---------------------------------------------------------------------------
__global__ __launch_bounds__(256, 1) void hr_sim(
    const float* __restrict__ sp0p, const float* __restrict__ sg0p,
    const float* __restrict__ W1,   const float* __restrict__ b1,
    const float* __restrict__ b2,   const float* __restrict__ b3,
    const float* __restrict__ rs1,  const float* __restrict__ rs2,
    const _Float16* __restrict__ w2p, const _Float16* __restrict__ w3p,
    float* __restrict__ out)
{
    extern __shared__ __align__(32) char smem[];
    float4*   w1l = (float4*)smem;              //  512 float4 = 8 KB
    _Float16* h1  = (_Float16*)(w1l + NB);      // 32768 halfs (64 KB) packed-A
    _Float16* h2  = h1 + 32768;                 // 32768 halfs (64 KB) packed-A
    float*    b2l = (float*)(h2 + 32768);       //  512 floats = 2 KB
    float*    spS = b2l + NB;                   //   64
    float*    sgS = spS + TM;                   //   64
    float*    dS  = sgS + TM;                   //   64 x 2 deltas

    const int tid  = threadIdx.x;
    const int lane = tid & 31;
    const int wv   = tid >> 5;
    const int half = lane >> 4;
    const int g0   = blockIdx.x * TM;
    const int nt0  = wv << 2;             // this wave's 64-column slice of H2

    const float sp0 = sp0p[0], sg0 = sg0p[0];
    float lp = 0.f, lg = 0.f, pnl = 0.f, sp = sp0, sg = sg0;

    // Stage W1+b1 (plain k order) and b2 into LDS once.
    for (int k = tid; k < NB; k += 256) {
        w1l[k] = make_float4(W1[k], W1[NB + k], W1[2 * NB + k], b1[k]);
        b2l[k] = b2[k];
    }
    if (tid < TM) { spS[tid] = sp0; sgS[tid] = sg0; }

    const v16h* w2v = (const v16h*)w2p;
    const v16h* w3v = (const v16h*)w3p;
    const v16h* h1v = (const v16h*)h1;
    const v16h* h2v = (const v16h*)h2;

    // Per-thread constants for the layer-1 packed-A index algebra:
    //   p = tid + 256*i  =>  e = tid&15 (const), mlow = tid>>4 (const),
    //   kt = (i>>1)&15, half = i&1, mt = i>>5,
    //   k  = kt*32 + half*8 + kbase(e)
    const int e1    = tid & 15;
    const int j1    = e1 >> 1;
    const int kbase = ((j1 >> 2) << 4) | ((j1 & 3) << 1) | (e1 & 1);
    const int mlow  = tid >> 4;

    __syncthreads();

    for (int step = 0; step < NSTEPS; ++step) {
        const float t = (float)step * DT_F;

        // Hoist this step's shocks: issue loads now, consume after the GEMM.
        float r1s = 0.f, r2s = 0.f;
        if (tid < TM) {
            size_t ofs = (size_t)step * (size_t)BATCH + (size_t)(g0 + tid);
            r1s = rs1[ofs];
            r2s = rs2[ofs];
        }
        // Warm the cache for this wave's first W2 k-chunk.
        #pragma unroll
        for (int n4 = 0; n4 < 4; ++n4)
            __builtin_prefetch((const char*)&w2v[(((nt0 + n4) << 4)) * 32 + lane], 0, 0);

        // ---- Layer 1 (VALU, LDS-only): h1 = relu(x @ W1 + b1), packed-A order
        {
            float spv[MT4], sgv[MT4];
            #pragma unroll
            for (int mt = 0; mt < MT4; ++mt) {
                int m = (mt << 4) | mlow;
                spv[mt] = spS[m];
                sgv[mt] = sgS[m];
            }
            #pragma unroll 4
            for (int ii = 0; ii < 32; ++ii) {
                int kt = ii >> 1, hf = ii & 1;
                int k  = (kt << 5) + (hf << 3) + kbase;
                float4 w = w1l[k];
                float wt = fmaf(w.z, t, w.w);
                int pbase = (((kt << 5) + mlow + (hf << 4)) << 4) | e1;
                #pragma unroll
                for (int mt = 0; mt < MT4; ++mt) {
                    float h = fmaf(w.x, spv[mt], fmaf(w.y, sgv[mt], wt));
                    h1[(mt << 13) + pbase] = (_Float16)fmaxf(h, 0.0f);
                }
            }
        }
        __syncthreads();

        // ---- Layer 2 (WMMA): H2 = relu(H1 @ W2 + b2), 64x512 per block
        v8f acc[MT4][4];
        {
            v8f z = {0.f,0.f,0.f,0.f,0.f,0.f,0.f,0.f};
            #pragma unroll
            for (int a = 0; a < MT4; ++a)
                #pragma unroll
                for (int b = 0; b < 4; ++b) acc[a][b] = z;
        }
        for (int kt = 0; kt < KT16; ++kt) {
            v16h bf[4];
            #pragma unroll
            for (int n4 = 0; n4 < 4; ++n4)
                bf[n4] = w2v[(((nt0 + n4) << 4) | kt) * 32 + lane];
            #pragma unroll
            for (int mt = 0; mt < MT4; ++mt) {
                v16h af = h1v[((mt << 4) | kt) * 32 + lane];
                #pragma unroll
                for (int n4 = 0; n4 < 4; ++n4)
                    acc[mt][n4] = __builtin_amdgcn_wmma_f32_16x16x32_f16(
                        false, af, false, bf[n4], (short)0, acc[mt][n4],
                        false, false);
            }
        }
        // bias + relu, store H2 directly in packed-A order (input to layer 3)
        #pragma unroll
        for (int mt = 0; mt < MT4; ++mt) {
            #pragma unroll
            for (int n4 = 0; n4 < 4; ++n4) {
                int ncol = ((nt0 + n4) << 4) | (lane & 15);   // D-layout column
                float bias = b2l[ncol];
                int ktp = ncol >> 5;
                int off = ncol & 31;
                int hh  = (off >> 3) & 1;
                int j   = ((off >> 4) << 2) | ((off >> 1) & 3);
                int e   = (j << 1) | (off & 1);
                int base = (((mt << 4) | ktp) * 32 + (hh << 4)) * 16 + e;
                #pragma unroll
                for (int r = 0; r < 8; ++r) {
                    int mrow = r + (half << 3);               // m & 15 (D layout)
                    float v = fmaxf(acc[mt][n4][r] + bias, 0.0f);
                    h2[base + mrow * 16] = (_Float16)v;
                }
            }
        }
        __syncthreads();

        // ---- Layer 3 (WMMA, N padded to 16): deltas = H2 @ W3p + b3
        if (wv < MT4) {
            v8f a3 = {0.f,0.f,0.f,0.f,0.f,0.f,0.f,0.f};
            const int mt = wv;
            #pragma unroll
            for (int kt = 0; kt < KT16; ++kt) {
                v16h af = h2v[((mt << 4) | kt) * 32 + lane];
                v16h bf = w3v[(kt << 5) + lane];
                a3 = __builtin_amdgcn_wmma_f32_16x16x32_f16(
                        false, af, false, bf, (short)0, a3, false, false);
            }
            int n = lane & 15;
            if (n < 2) {
                float bb = b3[n];
                #pragma unroll
                for (int r = 0; r < 8; ++r) {
                    int m = (mt << 4) + r + (half << 3);
                    dS[m * 2 + n] = a3[r] + bb;
                }
            }
        }
        __syncthreads();

        // ---- Path update (threads 0..63 own one path each, state in regs)
        if (tid < TM) {
            float wg = 0.8f * r1s + 0.6f * r2s;     // rho*w1 + sqrt(1-rho^2)*r2
            lp += DRIFT_P + 0.35f * r1s;
            lg += DRIFT_G + 0.60f * wg;
            float nsp = sp0 * __expf(lp);
            float nsg = sg0 * __expf(lg);
            pnl += dS[tid * 2 + 0] * (nsp - sp) + dS[tid * 2 + 1] * (nsg - sg);
            sp = nsp; sg = nsg;
            spS[tid] = sp; sgS[tid] = sg;
        }
        __syncthreads();
    }

    if (tid < TM) {
        // payoff = PSI * max(PHI*(sp - H_RATE*sg - K), 0) with PSI=-1, PHI=1
        out[g0 + tid] = pnl - fmaxf(sp - 2.35f * sg - 15.0f, 0.0f);
    }
}

// ---------------------------------------------------------------------------
extern "C" void kernel_launch(void* const* d_in, const int* in_sizes, int n_in,
                              void* d_out, int out_size, void* d_ws, size_t ws_size,
                              hipStream_t stream)
{
    const float* sp0 = (const float*)d_in[0];
    const float* sg0 = (const float*)d_in[1];
    const float* W1  = (const float*)d_in[2];
    const float* b1  = (const float*)d_in[3];
    const float* W2  = (const float*)d_in[4];
    const float* b2  = (const float*)d_in[5];
    const float* W3  = (const float*)d_in[6];
    const float* b3  = (const float*)d_in[7];
    const float* rs1 = (const float*)d_in[8];
    const float* rs2 = (const float*)d_in[9];
    float* out = (float*)d_out;

    // workspace: [0, 512KB) W2 packed f16 ; [512KB, 528KB) W3 packed f16
    _Float16* w2p = (_Float16*)d_ws;
    _Float16* w3p = (_Float16*)((char*)d_ws + 524288);

    hr_pack<<<1024, 256, 0, stream>>>(W2, W3, w2p, w3p);

    size_t lds = (size_t)NB * sizeof(float4)                  // w1l   8 KB
               + (size_t)(32768 + 32768) * sizeof(_Float16)   // h1+h2 128 KB
               + (size_t)NB * sizeof(float)                   // b2l   2 KB
               + (size_t)(TM + TM + 2 * TM) * sizeof(float);  // state
    hr_sim<<<BATCH / TM, 256, lds, stream>>>(sp0, sg0, W1, b1, b2, b3,
                                             rs1, rs2, w2p, w3p, out);
}
